// CapsuleLayer_16363825397895
// MI455X (gfx1250) — matrix operand order repaired and checked by reading, not compiled, not run on hardware
//
#include <hip/hip_runtime.h>

#define B_  32
#define N_  2048
#define D_  32
#define C_  64
#define V_  32
#define CV_ 2048

typedef __attribute__((ext_vector_type(16))) __bf16       v16bf;
typedef __attribute__((ext_vector_type(2)))  __bf16       v2bf;
typedef __attribute__((ext_vector_type(8)))  float        v8f;
typedef __attribute__((ext_vector_type(4)))  unsigned int u32x4;

// native f32 -> bf16 (v_cvt_pk_bf16_f32 codegen confirmed on gfx1250)
__device__ __forceinline__ unsigned short f2bf(float f) {
  return __builtin_bit_cast(unsigned short, (__bf16)f);
}

union FragA { v16bf v; u32x4 q[2]; };
union Pack8 { v2bf p[4]; u32x4 u; };

// ---------------------------------------------------------------------------
// K1: u_hat[b,n,cv] = sum_d X[b,n,d] * W[n,cv,d]   (bf16 WMMA, store bf16)
// u_hat stored as [N][CV][B] bf16 so routing passes get b contiguous in lanes.
// One block per n; 8 waves each own 16 cv-tiles of 16 columns; 2 M-tiles.
// ---------------------------------------------------------------------------
__global__ __launch_bounds__(256) void caps_gemm_uhat(
    const float* __restrict__ X,        // [B, N, D]
    const float* __restrict__ W,        // [N, C, V, D] (leading 1 dropped)
    unsigned short* __restrict__ UH)    // [N, CV, B] bf16
{
  __shared__ unsigned short Asm[B_ * D_];   // input[:, n, :] as bf16, [b][d]
  const int n    = blockIdx.x;
  const int tid  = threadIdx.x;
  const int wave = tid >> 5;
  const int lane = tid & 31;

  for (int i = tid; i < B_ * D_; i += 256) {
    int b = i >> 5, d = i & 31;
    Asm[i] = f2bf(X[((size_t)b * N_ + n) * D_ + d]);
  }
  __syncthreads();

  // A fragments (ISA 16-bit A 16x32 layout): lane l -> row (l&15)+16*mt,
  // K = { kh*8..kh*8+7 , 16+kh*8..16+kh*8+7 }, kh = l>>4.
  // Each K-half is 16 contiguous bytes in LDS -> two ds_load_b128.
  const int kh = lane >> 4;
  FragA afrag[2];
  #pragma unroll
  for (int mt = 0; mt < 2; ++mt) {
    const u32x4* q = (const u32x4*)&Asm[((lane & 15) + 16 * mt) * D_];
    afrag[mt].q[0] = q[kh];
    afrag[mt].q[1] = q[2 + kh];
  }

  const float* Wn = W + (size_t)n * (C_ * V_ * D_);
  for (int it = 0; it < 16; ++it) {
    const int cv0 = (wave + 8 * it) * 16;
    // B fragment: lane l -> column cv0+(l&15), K = { kh*8..kh*8+7 , 16+kh*8..16+kh*8+7 } (fp32)
    const float* wp = Wn + (size_t)(cv0 + (lane & 15)) * D_ + kh * 8;
    if (it + 1 < 16) __builtin_prefetch(wp + 128 * D_, 0, 1);  // next W tile
    FragA bfrag;
    {
      Pack8 lo, hi;
      #pragma unroll
      for (int r = 0; r < 4; ++r) {
        lo.p[r] = (v2bf){(__bf16)wp[2 * r],     (__bf16)wp[2 * r + 1]};
        hi.p[r] = (v2bf){(__bf16)wp[16 + 2 * r], (__bf16)wp[16 + 2 * r + 1]};
      }
      bfrag.q[0] = lo.u;
      bfrag.q[1] = hi.u;
    }

    v8f acc0 = {}, acc1 = {};
    acc0 = __builtin_amdgcn_wmma_f32_16x16x32_bf16(false, afrag[0].v, false, bfrag.v,
                                                   (short)0, acc0, false, false);
    acc1 = __builtin_amdgcn_wmma_f32_16x16x32_bf16(false, afrag[1].v, false, bfrag.v,
                                                   (short)0, acc1, false, false);

    // D layout: VGPR r holds (M = r + 8*(lane>=16), N = lane&15).
    // In [N][CV][B] the 8 M's per lane are contiguous -> one 16B bf16 store.
    const size_t base =
        ((size_t)n * CV_ + cv0 + (lane & 15)) * B_ + 8 * (lane >> 4);
    Pack8 o0, o1;
    #pragma unroll
    for (int r = 0; r < 4; ++r) {
      o0.p[r] = (v2bf){(__bf16)acc0[2 * r], (__bf16)acc0[2 * r + 1]};
      o1.p[r] = (v2bf){(__bf16)acc1[2 * r], (__bf16)acc1[2 * r + 1]};
    }
    *(u32x4*)(UH + base)      = o0.u;   // M-tile 0 (m 0..15)
    *(u32x4*)(UH + base + 16) = o1.u;   // M-tile 1 (m 16..31)
  }
}

// ---------------------------------------------------------------------------
// K2: one routing pass. lane = b. phase 0: r uniform (softmax of zeros).
// phase 1: b2 = u_hat.v1 (store), r=softmax(b2). phase 2: b3 = b2 + u_hat.v2.
// u_hat read through __bf16 loads so widening folds into v_fma_mix / v_cvt.
// s accumulated in 256KB LDS tile [CV][B] via ds_add_f32, flushed atomically.
// ---------------------------------------------------------------------------
__global__ __launch_bounds__(256) void caps_route_pass(
    const __bf16* __restrict__ UH,           // [N][CV][B] bf16
    float* __restrict__ Bacc,                // [N][C][B]
    const float* __restrict__ Vprev,         // [CV][B]
    float* __restrict__ Sout,                // [CV][B]
    int phase)
{
  extern __shared__ float s_local[];         // CV_*B_ floats = 256 KB
  const int tid  = threadIdx.x;
  const int wave = tid >> 5;
  const int lane = tid & 31;

  for (int i = tid; i < CV_ * B_; i += 256) s_local[i] = 0.0f;
  __syncthreads();

  const int npw = N_ / (gridDim.x * 8);      // grid 256 -> 1 n per wave
  for (int ni = 0; ni < npw; ++ni) {
    const int n = (blockIdx.x * 8 + wave) * npw + ni;
    const __bf16* uhn = UH + (size_t)n * CV_ * B_ + lane;

    float a[C_];
    if (phase != 0) {
      // agreement a[c] = sum_v u_hat[b,n,c,v] * v_prev[b,c,v]
      #pragma unroll
      for (int c = 0; c < C_; ++c) {
        const __bf16* up = uhn + (size_t)c * V_ * B_;
        const float* vp = Vprev + c * V_ * B_ + lane;
        float acc0 = 0.f, acc1 = 0.f;
        for (int v = 0; v < V_; v += 2) {
          acc0 += (float)up[(size_t)v * B_]       * vp[v * B_];
          acc1 += (float)up[(size_t)(v + 1) * B_] * vp[(v + 1) * B_];
        }
        a[c] = acc0 + acc1;
      }
      // routing logits + softmax over c (all in VGPRs)
      const size_t boff = ((size_t)n * C_) * B_ + lane;
      float m = -3.4e38f;
      #pragma unroll
      for (int c = 0; c < C_; ++c) {
        float bp = (phase == 2) ? Bacc[boff + (size_t)c * B_] : 0.f;
        a[c] += bp;
        if (phase == 1) Bacc[boff + (size_t)c * B_] = a[c];
        m = fmaxf(m, a[c]);
      }
      float sum = 0.f;
      #pragma unroll
      for (int c = 0; c < C_; ++c) { a[c] = __expf(a[c] - m); sum += a[c]; }
      const float inv = 1.f / sum;
      #pragma unroll
      for (int c = 0; c < C_; ++c) a[c] *= inv;
    }

    // s[b,c,v] += r[b,n,c] * u_hat[b,n,c,v]  (bank-conflict-free ds_add_f32)
    #pragma unroll
    for (int c = 0; c < C_; ++c) {
      const float rc = (phase == 0) ? (1.0f / C_) : a[c];
      const __bf16* up = uhn + (size_t)c * V_ * B_;
      float* sl = s_local + c * V_ * B_ + lane;
      for (int v = 0; v < V_; ++v)
        atomicAdd(sl + v * B_, rc * (float)up[(size_t)v * B_]);
    }
  }

  __syncthreads();
  for (int i = tid; i < CV_ * B_; i += 256) atomicAdd(&Sout[i], s_local[i]);
}

// ---------------------------------------------------------------------------
// K3: squash. final=1 writes d_out as [B,C,V]; else Vws as [CV][B].
// ---------------------------------------------------------------------------
__global__ __launch_bounds__(256) void caps_squash(
    const float* __restrict__ S, float* __restrict__ Vws,
    float* __restrict__ Out, int final_round)
{
  const int idx = blockIdx.x * blockDim.x + threadIdx.x;
  if (idx >= B_ * C_) return;
  const int b = idx / C_, c = idx % C_;
  float sv[V_];
  float sn = 0.f;
  #pragma unroll
  for (int v = 0; v < V_; ++v) {
    float t = S[(size_t)(c * V_ + v) * B_ + b];
    sv[v] = t;
    sn += t * t;
  }
  const float scale = (sn / (1.f + sn)) * rsqrtf(sn + 1e-9f);
  if (final_round) {
    #pragma unroll
    for (int v = 0; v < V_; ++v)
      Out[((size_t)b * C_ + c) * V_ + v] = sv[v] * scale;
  } else {
    #pragma unroll
    for (int v = 0; v < V_; ++v)
      Vws[(size_t)(c * V_ + v) * B_ + b] = sv[v] * scale;
  }
}

__global__ void caps_zero(float* __restrict__ p, int nElem) {
  int i = blockIdx.x * blockDim.x + threadIdx.x;
  if (i < nElem) p[i] = 0.f;
}

// ---------------------------------------------------------------------------
extern "C" void kernel_launch(void* const* d_in, const int* in_sizes, int n_in,
                              void* d_out, int out_size, void* d_ws, size_t ws_size,
                              hipStream_t stream) {
  (void)in_sizes; (void)n_in; (void)out_size; (void)ws_size;
  const float* X = (const float*)d_in[0];   // [B,N,D] fp32
  const float* W = (const float*)d_in[1];   // [1,N,C,V,D] fp32
  float* Out = (float*)d_out;               // [B,C,V] fp32

  char* ws = (char*)d_ws;
  unsigned short* UH = (unsigned short*)ws;                        // 256 MB bf16
  float* Bacc = (float*)(ws + (size_t)268435456);                  // 16 MB
  float* S    = (float*)(ws + (size_t)268435456 + 16777216);       // 256 KB
  float* Vws  = (float*)(ws + (size_t)268435456 + 16777216 + 262144); // 256 KB

  // allow 256 KB dynamic LDS for the routing pass (deterministic, capture-safe)
  (void)hipFuncSetAttribute((const void*)caps_route_pass,
                            hipFuncAttributeMaxDynamicSharedMemorySize,
                            CV_ * B_ * (int)sizeof(float));

  caps_gemm_uhat<<<N_, 256, 0, stream>>>(X, W, UH);

  for (int p = 0; p < 3; ++p) {
    caps_zero<<<(CV_ * B_ + 255) / 256, 256, 0, stream>>>(S, CV_ * B_);
    caps_route_pass<<<256, 256, CV_ * B_ * sizeof(float), stream>>>(
        (const __bf16*)UH, Bacc, Vws, S, p);
    caps_squash<<<(B_ * C_ + 255) / 256, 256, 0, stream>>>(S, Vws, Out, p == 2);
  }
}
